// VishwamaiAttention_27101243638359
// MI455X (gfx1250) — compile-verified, matrix operand order checked by reading, not run on hardware
//
#include <hip/hip_runtime.h>
#include <hip/hip_bf16.h>

// ---------------------------------------------------------------------------
// VishwamaiAttention for MI455X (gfx1250, wave32, WMMA + async LDS copies)
//
//   1) f32 -> bf16 convert of hidden_states; f32 -> bf16 transpose of weights
//   2) Q/K/V projections: bf16 WMMA GEMM, double-buffered LDS fed by
//      global_load_async_to_lds_b128 (ASYNCcnt), 64x64 per-wave tile
//   3) per-token head-mixing attention (RoPE fused, LDS k/v, VALU softmax)
//   4) output projection -> f32 d_out
// ---------------------------------------------------------------------------

typedef __attribute__((ext_vector_type(16))) __bf16 v16bf;
typedef __attribute__((ext_vector_type(8)))  __bf16 v8bf;
typedef __attribute__((ext_vector_type(4)))  __bf16 v4bf;
typedef __attribute__((ext_vector_type(8)))  float  v8f;
typedef __attribute__((ext_vector_type(4)))  float  v4f;

#define HSIZE 4096
#define KVD   1024
#define NH    32
#define NKV   8
#define DH    128
#define SEQ   4096
#define NBATCH 2
#define MROWS (NBATCH*SEQ)   // 8192

#define BM 256
#define BN 128
#define BK 32

// ---- CDNA5 async memory->LDS copy (ASYNCcnt), inline asm for portability ---
__device__ __forceinline__ void async_ld_b128(void* ldsDst, const void* gsrc) {
  // generic LDS pointer: addr[31:0] is the LDS byte offset (ISA 10.2 aperture)
  uint32_t off = (uint32_t)(uintptr_t)ldsDst;
  asm volatile("global_load_async_to_lds_b128 %0, %1, off"
               :: "v"(off), "v"(gsrc) : "memory");
}
__device__ __forceinline__ void wait_async0() {
  asm volatile("s_wait_asynccnt 0x0" ::: "memory");
}

// ---------------- f32 -> bf16 row-major convert ----------------
__global__ void k_cvt_bf16(const float* __restrict__ src,
                           __bf16* __restrict__ dst, int n4) {
  int i = blockIdx.x * blockDim.x + threadIdx.x;
  if (i >= n4) return;
  v4f f = *(const v4f*)(src + 4ll * i);
  v4bf b;
  b.x = (__bf16)f.x; b.y = (__bf16)f.y; b.z = (__bf16)f.z; b.w = (__bf16)f.w;
  *(v4bf*)(dst + 4ll * i) = b;
}

// ---------------- f32 [K][N] -> bf16 [N][K] transpose ----------------
__global__ void k_cvt_t(const float* __restrict__ src,
                        __bf16* __restrict__ dst, int K, int N) {
  __shared__ float tile[32][33];
  const int k0 = blockIdx.y * 32, n0 = blockIdx.x * 32;
  const int x = threadIdx.x, y = threadIdx.y;       // block (32,8)
  #pragma unroll
  for (int j = 0; j < 32; j += 8)
    tile[y + j][x] = src[(size_t)(k0 + y + j) * N + n0 + x];
  __syncthreads();
  #pragma unroll
  for (int j = 0; j < 32; j += 8)
    dst[(size_t)(n0 + y + j) * K + k0 + x] = (__bf16)tile[x][y + j];
}

// ---------------- bf16 WMMA GEMM:  C[M][N] = A[M][K] * Bt[N][K]^T ----------
// Block tile 256x128, BK=32, 256 threads (8 waves), wave tile 64x64,
// double-buffered LDS filled by async global->LDS copies.
__global__ __launch_bounds__(256)
void k_gemm_bf16(const __bf16* __restrict__ A, const __bf16* __restrict__ Bt,
                 float* __restrict__ C, int M, int N, int K) {
  __shared__ __align__(16) __bf16 As[2][BM * BK];   // 2 x 16 KB
  __shared__ __align__(16) __bf16 Bs[2][BN * BK];   // 2 x  8 KB

  const int tid  = threadIdx.x;
  const int lane = tid & 31;
  const int wave = tid >> 5;
  const int wm   = (wave & 3) * 64;   // wave M offset inside block tile
  const int wn   = (wave >> 2) * 64;  // wave N offset inside block tile
  const int mBase = blockIdx.y * BM;
  const int nBase = blockIdx.x * BN;
  const int lh = lane & 15;
  const int kg = (lane < 16) ? 0 : 8; // K sub-group per ISA 16-bit A layout

  const __bf16* Ablk = A  + (size_t)mBase * K;
  const __bf16* Bblk = Bt + (size_t)nBase * K;

  v8f acc[4][4];
  #pragma unroll
  for (int i = 0; i < 4; ++i)
    #pragma unroll
    for (int j = 0; j < 4; ++j)
      acc[i][j] = {};

  // async prefetch of one K-tile pair into LDS buffer `buf`
  auto prefetch = [&](int buf, int kt) {
    #pragma unroll
    for (int p = 0; p < 4; ++p) {                 // A: 256x32 bf16 = 16 KB
      int idx = p * 256 + tid;
      int r = idx >> 2, c = (idx & 3) * 8;
      async_ld_b128(&As[buf][r * BK + c], Ablk + (size_t)r * K + kt + c);
    }
    #pragma unroll
    for (int p = 0; p < 2; ++p) {                 // B: 128x32 bf16 = 8 KB
      int idx = p * 256 + tid;
      int r = idx >> 2, c = (idx & 3) * 8;
      async_ld_b128(&Bs[buf][r * BK + c], Bblk + (size_t)r * K + kt + c);
    }
  };

  prefetch(0, 0);
  int buf = 0;
  for (int kt = 0; kt < K; kt += BK, buf ^= 1) {
    wait_async0();          // my async writes into As/Bs[buf] are complete
    __syncthreads();        // everyone's are; prior reads of buf^1 are done
    if (kt + BK < K) prefetch(buf ^ 1, kt + BK);

    v16bf af[4], bfv[4];
    #pragma unroll
    for (int i = 0; i < 4; ++i) {
      const __bf16* pa = &As[buf][(wm + i * 16 + lh) * BK + kg];
      v8bf lo = *(const v8bf*)pa;          // K = kg .. kg+7
      v8bf hi = *(const v8bf*)(pa + 16);   // K = 16+kg .. 16+kg+7
      af[i] = __builtin_shufflevector(lo, hi, 0,1,2,3,4,5,6,7,
                                              8,9,10,11,12,13,14,15);
    }
    #pragma unroll
    for (int j = 0; j < 4; ++j) {
      const __bf16* pb = &Bs[buf][(wn + j * 16 + lh) * BK + kg];
      v8bf lo = *(const v8bf*)pb;
      v8bf hi = *(const v8bf*)(pb + 16);
      bfv[j] = __builtin_shufflevector(lo, hi, 0,1,2,3,4,5,6,7,
                                               8,9,10,11,12,13,14,15);
    }
    #pragma unroll
    for (int i = 0; i < 4; ++i)
      #pragma unroll
      for (int j = 0; j < 4; ++j)
        acc[i][j] = __builtin_amdgcn_wmma_f32_16x16x32_bf16(
            false, af[i], false, bfv[j], (short)0, acc[i][j], false, false);
  }

  // Epilogue per ISA 32-bit C/D layout: VGPR r -> M = r (+8 for lanes 16-31)
  const int rowAdd = (lane < 16) ? 0 : 8;
  #pragma unroll
  for (int i = 0; i < 4; ++i)
    #pragma unroll
    for (int j = 0; j < 4; ++j)
      #pragma unroll
      for (int r = 0; r < 8; ++r) {
        int row = mBase + wm + i * 16 + rowAdd + r;
        int col = nBase + wn + j * 16 + lh;
        C[(size_t)row * N + col] = acc[i][j][r];
      }
}

// ---------------- per-token head-mixing attention with fused RoPE ----------
// grid = MROWS tokens, 128 threads: thread = (h = tid/4, d-quarter = tid%4)
__global__ __launch_bounds__(128)
void k_attn(const float* __restrict__ Q, const float* __restrict__ Kb,
            const float* __restrict__ Vb, __bf16* __restrict__ O) {
  __shared__ float ks[NKV * DH];
  __shared__ float vs[NKV * DH];
  const int token = blockIdx.x;
  const int s     = token & (SEQ - 1);
  const int tid   = threadIdx.x;
  const float fpos = (float)s;

  // stage K (RoPE'd) and V into LDS
  const size_t kvoff = (size_t)token * (NKV * DH);
  for (int e = tid; e < NKV * DH; e += 128) {
    int d  = e & (DH - 1);
    int kh = e >> 7;
    float kx = Kb[kvoff + e];
    int pd   = (d < 64) ? d + 64 : d - 64;
    float px = Kb[kvoff + kh * DH + pd];
    float rot = (d < 64) ? -px : px;
    int   i   = d & 63;
    float ex  = (float)(i & ~1) * (1.0f / 64.0f);   // 2*floor(i/2)/half
    float freq = fpos * __powf(10000.0f, ex);       // reference's +exponent
    float c, sn; __sincosf(freq, &sn, &c);
    ks[e] = kx * c + rot * sn;
    vs[e] = Vb[kvoff + e];
  }
  __syncthreads();

  const int h     = tid >> 2;
  const int dbase = (tid & 3) * 32;
  const size_t qoff = (size_t)token * HSIZE + (size_t)h * DH;

  // load + RoPE my 32-element q slice
  float qr[32];
  #pragma unroll
  for (int dd = 0; dd < 32; ++dd) {
    int d    = dbase + dd;
    float qx = Q[qoff + d];
    int pd   = (d < 64) ? d + 64 : d - 64;
    float px = Q[qoff + pd];
    float rot = (d < 64) ? -px : px;
    int   i   = d & 63;
    float ex  = (float)(i & ~1) * (1.0f / 64.0f);
    float freq = fpos * __powf(10000.0f, ex);
    float c, sn; __sincosf(freq, &sn, &c);
    qr[dd] = qx * c + rot * sn;
  }

  // scores over 32 "key heads" (GQA: key head t -> kv head t/4)
  const float scale = 0.08838834764831845f;  // 1/sqrt(128)
  float sc[32];
  #pragma unroll
  for (int t = 0; t < 32; ++t) {
    const float* kr = &ks[(t >> 2) * DH + dbase];
    float p = 0.f;
    #pragma unroll
    for (int dd = 0; dd < 32; ++dd) p += qr[dd] * kr[dd];
    p += __shfl_xor(p, 1, 32);    // reduce across the 4 lanes sharing h
    p += __shfl_xor(p, 2, 32);
    sc[t] = p * scale;
  }

  // softmax (each thread holds the full 32-score row for its head)
  float m = sc[0];
  #pragma unroll
  for (int t = 1; t < 32; ++t) m = fmaxf(m, sc[t]);
  float sum = 0.f;
  #pragma unroll
  for (int t = 0; t < 32; ++t) { sc[t] = __expf(sc[t] - m); sum += sc[t]; }
  float inv = 1.0f / sum;
  #pragma unroll
  for (int t = 0; t < 32; ++t) sc[t] *= inv;

  // out = attn @ v  -> bf16 (feeds final WMMA GEMM)
  #pragma unroll
  for (int dd = 0; dd < 32; ++dd) {
    float o = 0.f;
    #pragma unroll
    for (int t = 0; t < 32; ++t) o += sc[t] * vs[(t >> 2) * DH + dbase + dd];
    O[qoff + dbase + dd] = (__bf16)o;
  }
}

// ---------------------------------------------------------------------------
extern "C" void kernel_launch(void* const* d_in, const int* in_sizes, int n_in,
                              void* d_out, int out_size, void* d_ws, size_t ws_size,
                              hipStream_t stream) {
  const float* hs = (const float*)d_in[0];
  const float* Wq = (const float*)d_in[1];
  const float* Wk = (const float*)d_in[2];
  const float* Wv = (const float*)d_in[3];
  const float* Wv_ = Wv; (void)Wv_;
  const float* Wo = (const float*)d_in[4];
  float* out = (float*)d_out;

  char* ws = (char*)d_ws;
  // workspace layout (bytes)
  __bf16* hsb = (__bf16*)(ws);                    //  64 MiB  (reused for attn out)
  __bf16* Wqt = (__bf16*)(ws +  67108864ll);      //  32 MiB
  __bf16* Wkt = (__bf16*)(ws + 100663296ll);      //   8 MiB
  __bf16* Wvt = (__bf16*)(ws + 109051904ll);      //   8 MiB
  __bf16* Wot = (__bf16*)(ws + 117440512ll);      //  32 MiB
  float*  Qf  = (float*) (ws + 150994944ll);      // 128 MiB
  float*  Kf  = (float*) (ws + 285212672ll);      //  32 MiB
  float*  Vf  = (float*) (ws + 318767104ll);      //  32 MiB  (total ~352 MiB)

  // 1) precision staging
  const int n4 = MROWS * HSIZE / 4;
  k_cvt_bf16<<<n4 / 256, 256, 0, stream>>>(hs, hsb, n4);
  dim3 tb(32, 8);
  k_cvt_t<<<dim3(HSIZE / 32, HSIZE / 32), tb, 0, stream>>>(Wq, Wqt, HSIZE, HSIZE);
  k_cvt_t<<<dim3(KVD   / 32, HSIZE / 32), tb, 0, stream>>>(Wk, Wkt, HSIZE, KVD);
  k_cvt_t<<<dim3(KVD   / 32, HSIZE / 32), tb, 0, stream>>>(Wv, Wvt, HSIZE, KVD);
  k_cvt_t<<<dim3(HSIZE / 32, HSIZE / 32), tb, 0, stream>>>(Wo, Wot, HSIZE, HSIZE);

  // 2) Q/K/V projections (WMMA bf16, f32 accumulate)
  k_gemm_bf16<<<dim3(HSIZE / BN, MROWS / BM), 256, 0, stream>>>(hsb, Wqt, Qf, MROWS, HSIZE, HSIZE);
  k_gemm_bf16<<<dim3(KVD   / BN, MROWS / BM), 256, 0, stream>>>(hsb, Wkt, Kf, MROWS, KVD,   HSIZE);
  k_gemm_bf16<<<dim3(KVD   / BN, MROWS / BM), 256, 0, stream>>>(hsb, Wvt, Vf, MROWS, KVD,   HSIZE);

  // 3) per-token attention (RoPE fused); writes bf16 into reused hsb buffer
  __bf16* attnb = hsb;
  k_attn<<<MROWS, 128, 0, stream>>>(Qf, Kf, Vf, attnb);

  // 4) output projection -> f32 d_out
  k_gemm_bf16<<<dim3(HSIZE / BN, MROWS / BM), 256, 0, stream>>>(attnb, Wot, out, MROWS, HSIZE, HSIZE);
}
